// MemoryAttention_29557964931681
// MI455X (gfx1250) — compile-verified, matrix operand order checked by reading, not run on hardware
//
#include <hip/hip_runtime.h>
#include <hip/hip_fp16.h>

// ---------------------------------------------------------------------------
// MemoryAttention (SAM2-style) for MI455X / gfx1250, wave32 + WMMA.
// D=256, H=1, L=4, S_cur=4096, S_mem=16384, DFF=2048, B=1.
// ~410 GFLOP, working set << 192MB L2  =>  compute-bound: f16 WMMA w/ f32
// accumulation; flash-attention kernel with LDS-staged K/V^T tiles loaded via
// global_load_async_to_lds_b128 (double-buffered, ASYNCcnt + barriers).
// ---------------------------------------------------------------------------

typedef __attribute__((ext_vector_type(16))) _Float16 v16h;
typedef __attribute__((ext_vector_type(8)))  float    v8f;

static constexpr int SC  = 4096;   // current tokens
static constexpr int SM  = 16384;  // memory tokens
static constexpr int DD  = 256;    // model dim (= head dim, H=1)
static constexpr int DFF = 2048;
static constexpr int LAY = 4;

// ---- fragment loader -------------------------------------------------------
// Per-lane layout for 16-bit A (16x32) operand (B via B^T == W rows):
//   lane<16 : row = rowbase+lane,    K = k0 + {0..7} and k0 + {16..23}
//   lane>=16: row = rowbase+lane-16, K = k0 + {8..15} and k0 + {24..31}
__device__ __forceinline__ v16h frag_ab(const _Float16* base, int ld,
                                        int rowbase, int k0) {
  int lane = threadIdx.x & 31;
  int r    = rowbase + (lane & 15);
  int koff = (lane >> 4) << 3;            // 0 or 8
  const _Float16* p = base + (size_t)r * ld + k0 + koff;
  v16h out;
  ((float4*)&out)[0] = ((const float4*)p)[0];        // K = koff + 0..7
  ((float4*)&out)[1] = ((const float4*)(p + 16))[0]; // K = koff + 16..23
  return out;
}

__device__ __forceinline__ v8f wmma16(v16h a, v16h b, v8f c) {
  return __builtin_amdgcn_wmma_f32_16x16x32_f16(false, a, false, b,
                                                (short)0, c, false, false);
}

// ---- async global -> LDS copy (CDNA5 ASYNCcnt path) ------------------------
__device__ __forceinline__ void async_b128(unsigned ldsoff, const void* gsrc) {
  unsigned long long ga = (unsigned long long)gsrc;
  asm volatile("global_load_async_to_lds_b128 %0, %1, off"
               :: "v"(ldsoff), "v"(ga) : "memory");
}
__device__ __forceinline__ void wait_async0() {
  asm volatile("s_wait_asynccnt 0" ::: "memory");
}
__device__ __forceinline__ unsigned lds_off(const void* p) {
  return (unsigned)(unsigned long long)p;   // low 32 bits = LDS byte offset
}

// ---- generic GEMM: C[M,N] = A[M,K] @ W[N,K]^T + bias -----------------------
// 32x64 tile per wave, register double-buffered K-loop.
// flags: 1 = ReLU, 2 = write f16 out, 4 = accumulate f32 into `accum`
__global__ __launch_bounds__(256) void gemm_wmma_kernel(
    const _Float16* __restrict__ A, const _Float16* __restrict__ W,
    const float* __restrict__ bias, _Float16* __restrict__ outh,
    float* __restrict__ accum, int M, int N, int K, int flags) {
  int lane  = threadIdx.x & 31;
  int wid   = (blockIdx.x * blockDim.x + threadIdx.x) >> 5;
  int tilesN = N >> 6;                        // 64-wide strips
  int tm = wid / tilesN, tn = wid % tilesN;
  int m0 = tm * 32, n0 = tn * 64;
  if (m0 >= M) return;
  int lr = lane & 15, hi = lane >> 4;

  v8f acc[2][4];
#pragma unroll
  for (int r = 0; r < 2; ++r)
#pragma unroll
    for (int j = 0; j < 4; ++j) acc[r][j] = {};

  v16h a0 = frag_ab(A, K, m0, 0);
  v16h a1 = frag_ab(A, K, m0 + 16, 0);
  v16h b[4];
#pragma unroll
  for (int j = 0; j < 4; ++j) b[j] = frag_ab(W, K, n0 + j * 16, 0);

  for (int k0 = 0;;) {
    int kn = k0 + 32;
    v16h a0n, a1n, bn[4];
    bool more = kn < K;
    if (more) {                                // issue next-slice loads early
      a0n = frag_ab(A, K, m0, kn);
      a1n = frag_ab(A, K, m0 + 16, kn);
#pragma unroll
      for (int j = 0; j < 4; ++j) bn[j] = frag_ab(W, K, n0 + j * 16, kn);
    }
#pragma unroll
    for (int j = 0; j < 4; ++j) {
      acc[0][j] = wmma16(a0, b[j], acc[0][j]);
      acc[1][j] = wmma16(a1, b[j], acc[1][j]);
    }
    if (!more) break;
    a0 = a0n; a1 = a1n;
#pragma unroll
    for (int j = 0; j < 4; ++j) b[j] = bn[j];
    k0 = kn;
  }
#pragma unroll
  for (int r = 0; r < 2; ++r)
#pragma unroll
    for (int j = 0; j < 4; ++j)
#pragma unroll
      for (int i = 0; i < 8; ++i) {
        int row = m0 + r * 16 + i + 8 * hi;    // C layout: vgpr i, half hi
        int col = n0 + j * 16 + lr;
        float v = acc[r][j][i] + (bias ? bias[col] : 0.f);
        if (flags & 1) v = fmaxf(v, 0.f);
        size_t idx = (size_t)row * N + col;
        if (flags & 2) outh[idx] = (_Float16)v;
        if (flags & 4) accum[idx] += v;
      }
}

// ---- fused flash attention -------------------------------------------------
// Block = 8 waves = 128 queries; all waves share the same key chunks, so K and
// V^T tiles are staged once per block in LDS via async copies, double-buffered.
__global__ __launch_bounds__(256) void attn_kernel(
    const _Float16* __restrict__ Q, const _Float16* __restrict__ Km,
    const _Float16* __restrict__ VT, _Float16* __restrict__ Out,
    int M, int Sk, float scale) {
  __shared__ _Float16 kls[2][32 * 256];       // K tile: 32 keys x 256 dims
  __shared__ _Float16 vtls[2][256 * 32];      // V^T tile: 256 dims x 32 keys
  __shared__ _Float16 pbuf[8][16 * 32];       // per-wave P tile

  int tid  = threadIdx.x;
  int lane = tid & 31;
  int wid  = tid >> 5;
  int m0   = (blockIdx.x * 8 + wid) * 16;     // grid sized exactly: always < M
  int lr = lane & 15, hi = lane >> 4;
  _Float16* pls = &pbuf[wid][0];

  // cooperative async prefetch of one (K, V^T) tile pair into buffer `buf`
  auto prefetch = [&](int buf, int kb) {
    const char* ksrc = (const char*)(Km + (size_t)kb * DD);   // contiguous 16KB
    unsigned kbase = lds_off(&kls[buf][0]);
#pragma unroll
    for (int i = 0; i < 4; ++i) {
      int c = i * 256 + tid;                  // 16B chunk id, 0..1023
      async_b128(kbase + c * 16, ksrc + c * 16);
    }
    unsigned vbase = lds_off(&vtls[buf][0]);
#pragma unroll
    for (int i = 0; i < 4; ++i) {
      int c = i * 256 + tid;
      int row = c >> 2, seg = c & 3;          // 256 rows x 64B, stride Sk*2
      const char* src = (const char*)(VT + (size_t)row * Sk + kb) + seg * 16;
      async_b128(vbase + c * 16, src);
    }
  };

  v16h q[8];                                  // Q rows, all 256 dims
#pragma unroll
  for (int c = 0; c < 8; ++c) q[c] = frag_ab(Q, DD, m0, c * 32);

  v8f oacc[16];                               // 16x256 f32 output accum
#pragma unroll
  for (int t = 0; t < 16; ++t) oacc[t] = {};
  float mrow[8], lrow[8];
#pragma unroll
  for (int i = 0; i < 8; ++i) { mrow[i] = -1e30f; lrow[i] = 0.f; }

  prefetch(0, 0);
  wait_async0();
  __syncthreads();

  int nIter = Sk >> 5;
  for (int it = 0; it < nIter; ++it) {
    int cur = it & 1;
    if (it + 1 < nIter) prefetch(cur ^ 1, (it + 1) * 32);
    const _Float16* kt = &kls[cur][0];
    const _Float16* vt = &vtls[cur][0];

    // scores S = Q @ K^T for 32 keys -> two 16x16 f32 tiles (from LDS)
    v8f s0 = {}, s1 = {};
#pragma unroll
    for (int c = 0; c < 8; ++c) {
      v16h b0 = frag_ab(kt, DD, 0, c * 32);
      v16h b1 = frag_ab(kt, DD, 16, c * 32);
      s0 = wmma16(q[c], b0, s0);
      s1 = wmma16(q[c], b1, s1);
    }
    // online softmax; a row's 16 values live in one vgpr across a 16-lane half
#pragma unroll
    for (int i = 0; i < 8; ++i) {
      float v0 = s0[i] * scale, v1 = s1[i] * scale;
      float mx = fmaxf(v0, v1);
#pragma unroll
      for (int m = 8; m >= 1; m >>= 1) mx = fmaxf(mx, __shfl_xor(mx, m, 32));
      float newm = fmaxf(mrow[i], mx);
      float corr = __expf(mrow[i] - newm);
      float p0 = __expf(v0 - newm), p1 = __expf(v1 - newm);
      float ps = p0 + p1;
#pragma unroll
      for (int m = 8; m >= 1; m >>= 1) ps += __shfl_xor(ps, m, 32);
      lrow[i] = lrow[i] * corr + ps;
      mrow[i] = newm;
#pragma unroll
      for (int t = 0; t < 16; ++t) oacc[t][i] = oacc[t][i] * corr;
      int prow = i + 8 * hi;                  // transpose P via LDS
      pls[prow * 32 + lr]      = (_Float16)p0;
      pls[prow * 32 + 16 + lr] = (_Float16)p1;
    }
    asm volatile("s_wait_dscnt 0" ::: "memory");  // same-wave cross-lane RAW
    v16h ap = frag_ab(pls, 32, 0, 0);         // P as A operand (16x32)
#pragma unroll
    for (int t = 0; t < 16; ++t) {            // O += P @ V  (V^T rows in LDS)
      v16h b = frag_ab(vt, 32, t * 16, 0);
      oacc[t] = wmma16(ap, b, oacc[t]);
    }
    wait_async0();                            // our prefetch landed
    __syncthreads();                          // all waves done with `cur`
  }
#pragma unroll
  for (int t = 0; t < 16; ++t)
#pragma unroll
    for (int i = 0; i < 8; ++i) {
      int row = m0 + i + 8 * hi;
      int col = t * 16 + lr;
      Out[(size_t)row * DD + col] = (_Float16)(oacc[t][i] / lrow[i]);
    }
}

// ---- LayerNorm: one wave per row of 256 ------------------------------------
__global__ __launch_bounds__(256) void layernorm_kernel(
    const float* __restrict__ x, const float* __restrict__ w,
    const float* __restrict__ b, _Float16* __restrict__ outh,
    float* __restrict__ outf, int S) {
  int lane = threadIdx.x & 31;
  int row  = blockIdx.x * 8 + (threadIdx.x >> 5);
  if (row >= S) return;
  const float* xr = x + (size_t)row * DD;
  float v[8], s = 0.f;
#pragma unroll
  for (int t = 0; t < 8; ++t) { v[t] = xr[t * 32 + lane]; s += v[t]; }
#pragma unroll
  for (int m = 16; m >= 1; m >>= 1) s += __shfl_xor(s, m, 32);
  float mu = s * (1.f / 256.f), q = 0.f;
#pragma unroll
  for (int t = 0; t < 8; ++t) { float d = v[t] - mu; q += d * d; }
#pragma unroll
  for (int m = 16; m >= 1; m >>= 1) q += __shfl_xor(q, m, 32);
  float rstd = rsqrtf(q * (1.f / 256.f) + 1e-5f);
#pragma unroll
  for (int t = 0; t < 8; ++t) {
    int col = t * 32 + lane;
    float y = (v[t] - mu) * rstd * w[col] + b[col];
    if (outh) outh[(size_t)row * DD + col] = (_Float16)y;
    if (outf) outf[(size_t)row * DD + col] = y;
  }
}

// ---- elementwise helpers ---------------------------------------------------
__global__ void axpy_f32_kernel(const float* a, const float* b, float s,
                                float* o, int n) {
  int i = blockIdx.x * 256 + threadIdx.x;
  if (i < n) o[i] = a[i] + s * b[i];
}
__global__ void axpy_f16_kernel(const float* a, const float* b, float s,
                                _Float16* o, int n) {
  int i = blockIdx.x * 256 + threadIdx.x;
  if (i < n) o[i] = (_Float16)(a[i] + s * b[i]);
}
__global__ void cvt_f16_kernel(const float* a, _Float16* o, int n) {
  int i = blockIdx.x * 256 + threadIdx.x;
  if (i < n) o[i] = (_Float16)a[i];
}
__global__ void transpose_f16_kernel(const _Float16* in, _Float16* out, int S) {
  int i = blockIdx.x * 256 + threadIdx.x;
  if (i < S * DD) { int s = i >> 8, d = i & 255; out[(size_t)d * S + s] = in[i]; }
}

// ---- host orchestration ----------------------------------------------------
static inline void gemm(const _Float16* A, const _Float16* W, const float* bias,
                        _Float16* outh, float* accum, int M, int N, int K,
                        int flags, hipStream_t st) {
  int waves = (M / 32) * (N / 64);
  gemm_wmma_kernel<<<(waves * 32 + 255) / 256, 256, 0, st>>>(
      A, W, bias, outh, accum, M, N, K, flags);
}

extern "C" void kernel_launch(void* const* d_in, const int* in_sizes, int n_in,
                              void* d_out, int out_size, void* d_ws, size_t ws_size,
                              hipStream_t stream) {
  (void)in_sizes; (void)n_in; (void)out_size; (void)ws_size;
  const float* curr       = (const float*)d_in[0];
  const float* memory     = (const float*)d_in[1];
  const float* curr_pos   = (const float*)d_in[2];
  const float* memory_pos = (const float*)d_in[3];
  const float* sa_w_qkv = (const float*)d_in[4];
  const float* sa_b_qkv = (const float*)d_in[5];
  const float* sa_w_o   = (const float*)d_in[6];
  const float* sa_b_o   = (const float*)d_in[7];
  const float* ca_w_qkv = (const float*)d_in[8];
  const float* ca_b_qkv = (const float*)d_in[9];
  const float* ca_w_o   = (const float*)d_in[10];
  const float* ca_b_o   = (const float*)d_in[11];
  const float* ln1_w = (const float*)d_in[12]; const float* ln1_b = (const float*)d_in[13];
  const float* ln2_w = (const float*)d_in[14]; const float* ln2_b = (const float*)d_in[15];
  const float* ln3_w = (const float*)d_in[16]; const float* ln3_b = (const float*)d_in[17];
  const float* mlp_w1 = (const float*)d_in[18]; const float* mlp_b1 = (const float*)d_in[19];
  const float* mlp_w2 = (const float*)d_in[20]; const float* mlp_b2 = (const float*)d_in[21];
  const float* norm_w = (const float*)d_in[22]; const float* norm_b = (const float*)d_in[23];

  size_t off = 0;
  auto take = [&](size_t bytes) {
    size_t o = off; off += (bytes + 255) & ~(size_t)255;
    return (char*)d_ws + o;
  };
  float*    xf    = (float*)   take((size_t)SC * DD * 4);   // residual stream
  _Float16* t2h   = (_Float16*)take((size_t)SC * DD * 2);   // LN output
  _Float16* qh    = (_Float16*)take((size_t)SC * DD * 2);
  _Float16* kh    = (_Float16*)take((size_t)SM * DD * 2);
  _Float16* vh    = (_Float16*)take((size_t)SM * DD * 2);
  _Float16* vth   = (_Float16*)take((size_t)SM * DD * 2);   // V transposed
  _Float16* attnh = (_Float16*)take((size_t)SC * DD * 2);
  _Float16* h1h   = (_Float16*)take((size_t)SC * DFF * 2);  // MLP hidden
  _Float16* memkh = (_Float16*)take((size_t)SM * DD * 2);   // memory + pos (keys)
  _Float16* memh  = (_Float16*)take((size_t)SM * DD * 2);   // memory (values)
  _Float16* wsaq  = (_Float16*)take((size_t)LAY * 3 * DD * DD * 2);
  _Float16* wcaq  = (_Float16*)take((size_t)LAY * 3 * DD * DD * 2);
  _Float16* wsao  = (_Float16*)take((size_t)LAY * DD * DD * 2);
  _Float16* wcao  = (_Float16*)take((size_t)LAY * DD * DD * 2);
  _Float16* wm1   = (_Float16*)take((size_t)LAY * DFF * DD * 2);
  _Float16* wm2   = (_Float16*)take((size_t)LAY * DD * DFF * 2);

  const int nC = SC * DD, nM = SM * DD;
  // weight conversion f32 -> f16
  cvt_f16_kernel<<<(LAY*3*DD*DD+255)/256,256,0,stream>>>(sa_w_qkv, wsaq, LAY*3*DD*DD);
  cvt_f16_kernel<<<(LAY*3*DD*DD+255)/256,256,0,stream>>>(ca_w_qkv, wcaq, LAY*3*DD*DD);
  cvt_f16_kernel<<<(LAY*DD*DD+255)/256,256,0,stream>>>(sa_w_o, wsao, LAY*DD*DD);
  cvt_f16_kernel<<<(LAY*DD*DD+255)/256,256,0,stream>>>(ca_w_o, wcao, LAY*DD*DD);
  cvt_f16_kernel<<<(LAY*DFF*DD+255)/256,256,0,stream>>>(mlp_w1, wm1, LAY*DFF*DD);
  cvt_f16_kernel<<<(LAY*DD*DFF+255)/256,256,0,stream>>>(mlp_w2, wm2, LAY*DD*DFF);
  // x = curr + 0.1*curr_pos ; mem_k = memory + memory_pos ; mem (f16 copies)
  axpy_f32_kernel<<<nC/256,256,0,stream>>>(curr, curr_pos, 0.1f, xf, nC);
  axpy_f16_kernel<<<nM/256,256,0,stream>>>(memory, memory_pos, 1.0f, memkh, nM);
  axpy_f16_kernel<<<nM/256,256,0,stream>>>(memory, memory, 0.0f, memh, nM);

  const float scale = 1.0f / 16.0f;  // 1/sqrt(256)
  for (int l = 0; l < LAY; ++l) {
    // ---- self-attention (pre-norm) ----
    layernorm_kernel<<<SC/8,256,0,stream>>>(xf, ln1_w+l*DD, ln1_b+l*DD, t2h, nullptr, SC);
    const _Float16* wq = wsaq + (size_t)l*3*DD*DD;
    const float*    bq = sa_b_qkv + (size_t)l*3*DD;
    gemm(t2h, wq,            bq,        qh, nullptr, SC, DD, DD, 2, stream);
    gemm(t2h, wq +   DD*DD,  bq +   DD, kh, nullptr, SC, DD, DD, 2, stream);
    gemm(t2h, wq + 2*DD*DD,  bq + 2*DD, vh, nullptr, SC, DD, DD, 2, stream);
    transpose_f16_kernel<<<nC/256,256,0,stream>>>(vh, vth, SC);
    attn_kernel<<<SC/128,256,0,stream>>>(qh, kh, vth, attnh, SC, SC, scale);
    gemm(attnh, wsao + (size_t)l*DD*DD, sa_b_o + l*DD, nullptr, xf, SC, DD, DD, 4, stream);

    // ---- cross-attention to memory (keys carry pos, values do not) ----
    layernorm_kernel<<<SC/8,256,0,stream>>>(xf, ln2_w+l*DD, ln2_b+l*DD, t2h, nullptr, SC);
    const _Float16* wcq = wcaq + (size_t)l*3*DD*DD;
    const float*    bcq = ca_b_qkv + (size_t)l*3*DD;
    gemm(t2h,   wcq,           bcq,        qh, nullptr, SC, DD, DD, 2, stream);
    gemm(memkh, wcq +   DD*DD, bcq +   DD, kh, nullptr, SM, DD, DD, 2, stream);
    gemm(memh,  wcq + 2*DD*DD, bcq + 2*DD, vh, nullptr, SM, DD, DD, 2, stream);
    transpose_f16_kernel<<<nM/256,256,0,stream>>>(vh, vth, SM);
    attn_kernel<<<SC/128,256,0,stream>>>(qh, kh, vth, attnh, SC, SM, scale);
    gemm(attnh, wcao + (size_t)l*DD*DD, ca_b_o + l*DD, nullptr, xf, SC, DD, DD, 4, stream);

    // ---- MLP (pre-norm, ReLU) ----
    layernorm_kernel<<<SC/8,256,0,stream>>>(xf, ln3_w+l*DD, ln3_b+l*DD, t2h, nullptr, SC);
    gemm(t2h, wm1 + (size_t)l*DFF*DD, mlp_b1 + (size_t)l*DFF, h1h, nullptr,
         SC, DFF, DD, 1 | 2, stream);
    gemm(h1h, wm2 + (size_t)l*DD*DFF, mlp_b2 + (size_t)l*DD, nullptr, xf,
         SC, DD, DFF, 4, stream);
  }
  // final LayerNorm -> f32 output
  layernorm_kernel<<<SC/8,256,0,stream>>>(xf, norm_w, norm_b, nullptr, (float*)d_out, SC);
}
